// PoincareMPNN_27685359190105
// MI455X (gfx1250) — matrix operand order repaired and checked by reading, not compiled, not run on hardware
//
#include <hip/hip_runtime.h>
#include <hip/hip_bf16.h>

#define DD    128
#define NN    512
#define BB    2
#define DEG   32
#define EE    (BB*NN*DEG)     // 32768
#define K3    384

typedef __attribute__((ext_vector_type(16))) _Float16 v16h;
typedef __attribute__((ext_vector_type(8)))  _Float16 h8v;
typedef __attribute__((ext_vector_type(8)))  float    v8f;
typedef __attribute__((ext_vector_type(4)))  float    f4v;

// ---------------- helpers ----------------
__device__ __forceinline__ float grp16_sum(float v) {
    v += __shfl_xor(v, 1, 16);
    v += __shfl_xor(v, 2, 16);
    v += __shfl_xor(v, 4, 16);
    v += __shfl_xor(v, 8, 16);
    return v;
}

// logmap0 scale factor: atanh(clip(norm))/clip(norm), input = squared norm
__device__ __forceinline__ float logfac(float s) {
    float n  = sqrtf(s);
    float nc = fminf(fmaxf(n, 1e-7f), 1.f - 1e-5f);
    return atanhf(nc) / nc;
}

__device__ __forceinline__ v16h load_frag_lds(const _Float16* p0, const _Float16* p1) {
    h8v lo = *(const h8v*)p0;
    h8v hi = *(const h8v*)p1;
    return __builtin_shufflevector(lo, hi, 0,1,2,3,4,5,6,7,8,9,10,11,12,13,14,15);
}

// ---------------- kernel 0: convert weights fp32 -> fp16 (B-matrix layout == row major [n][k]) ----
__global__ __launch_bounds__(256) void pmpnn_cvt_weights(
    const float* __restrict__ We, const float* __restrict__ Wn,
    const float* __restrict__ Wa, const float* __restrict__ Wp,
    _Float16* __restrict__ WeH, _Float16* __restrict__ WnH,
    _Float16* __restrict__ WaH, _Float16* __restrict__ WpH)
{
    int idx = blockIdx.x * 256 + threadIdx.x;
    if (idx < DD*K3)  WeH[idx] = (_Float16)We[idx];   // 128x384
    if (idx < DD*256) WnH[idx] = (_Float16)Wn[idx];   // 128x256
    if (idx < K3*DD)  WaH[idx] = (_Float16)Wa[idx];   // 384x128
    if (idx < DD*DD)  WpH[idx] = (_Float16)Wp[idx];   // 128x128
}

// ---------------- kernel 1: e -> e_new bulk copy (nontemporal, copy-bound) ----------------
__global__ __launch_bounds__(256) void pmpnn_copy_e(const f4v* __restrict__ src,
                                                    f4v* __restrict__ dst, long n4)
{
    long idx    = (long)blockIdx.x * blockDim.x + threadIdx.x;
    long stride = (long)gridDim.x * blockDim.x;
    for (; idx < n4; idx += stride) {
        f4v v = __builtin_nontemporal_load(&src[idx]);
        __builtin_nontemporal_store(v, &dst[idx]);
    }
}

// ---------------- kernel 2: adjacency -> ordered edge list (wave32 ballot compaction) ----------
__global__ __launch_bounds__(256) void pmpnn_build_edges(const unsigned char* __restrict__ adj,
                                                         int* __restrict__ jlist,
                                                         int* __restrict__ deg)
{
    int wid  = (int)((blockIdx.x * blockDim.x + threadIdx.x) >> 5);  // row = b*N+i, 0..1023
    int lane = threadIdx.x & 31;
    const unsigned char* rowp = adj + (size_t)wid * NN;
    int count = 0;
    for (int base = 0; base < NN; base += 32) {
        bool p = rowp[base + lane] != 0;
        unsigned mask = (unsigned)__ballot(p);
        if (p) {
            int off = count + __popc(mask & ((1u << lane) - 1u));
            if (off < DEG) jlist[wid * DEG + off] = base + lane;
        }
        count += __popc(mask);
    }
    if (lane == 0) deg[wid] = count;
}

// ---------------- kernel 3: edge model (16 edges / block, 8 waves, 12 WMMA each) ---------------
__global__ __launch_bounds__(256) void pmpnn_edge_kernel(
    const float* __restrict__ x, const float* __restrict__ e,
    const int* __restrict__ jlist,
    const _Float16* __restrict__ WeH,
    const float* __restrict__ b_edge,
    const float* __restrict__ W_ada, const float* __restrict__ b_ada,
    float* __restrict__ eupd, float* __restrict__ out_e)
{
    __shared__ __align__(16) _Float16 Ah[16 * 392];   // 16 x 384 edge_in (f16), pitch 392
    __shared__ __align__(16) float    Z[16 * 132];    // 16 x 128 z tile
    __shared__ float Sdist[16];

    const int tid  = threadIdx.x;
    const int g    = tid >> 4;          // edge slot 0..15
    const int l16  = tid & 15;
    const int eIdx = blockIdx.x * 16 + g;
    const int row  = eIdx >> 5;         // b*N+i in [0,1024)
    const int b    = row >> 9;
    const int j    = jlist[eIdx];

    const float* xi = x + (size_t)row * DD;
    const float* xj = x + ((size_t)(b << 9) + j) * DD;
    const float* ef = e + ((size_t)row * NN + j) * DD;

    // ---- phase 1: norms, dist, logmap0, stage A tile ----
    float vx[8], vy[8], ve[8];
    float sii = 0.f, sjj = 0.f, sdd = 0.f, see = 0.f;
#pragma unroll
    for (int t = 0; t < 8; ++t) {
        float a = xi[l16 * 8 + t], c = xj[l16 * 8 + t], w = ef[l16 * 8 + t];
        vx[t] = a; vy[t] = c; ve[t] = w;
        sii += a * a; sjj += c * c; see += w * w;
        float d0 = a - c; sdd += d0 * d0;
    }
    sii = grp16_sum(sii); sjj = grp16_sum(sjj);
    sdd = grp16_sum(sdd); see = grp16_sum(see);

    float den  = fmaxf((1.f - sii) * (1.f - sjj), 1e-12f);
    float arg  = fmaxf(1.f + 2.f * sdd / den, 1.f + 1e-7f);
    float dist = fminf(fmaxf(acoshf(arg), 1e-6f), 100.f);
    float sd   = dist / (1.f + expf(-dist));        // silu
    if (l16 == 0) Sdist[g] = sd;

    float fi = logfac(sii), fj = logfac(sjj), fe = logfac(see);
#pragma unroll
    for (int t = 0; t < 8; ++t) {
        int d = l16 * 8 + t;
        Ah[g * 392 + d]       = (_Float16)(vx[t] * fi);
        Ah[g * 392 + 128 + d] = (_Float16)(vy[t] * fj);
        Ah[g * 392 + 256 + d] = (_Float16)(ve[t] * fe);
    }
    __syncthreads();

    // ---- phase 2: z = edge_in @ W_edge^T, one 16x16 N-tile per wave, K=384 ----
    const int wave = tid >> 5, lane = tid & 31;
    const int n0 = wave * 16, arow = lane & 15, hsel = lane >> 4;
    v8f acc = {};
#pragma unroll
    for (int kb = 0; kb < 12; ++kb) {
        int ka = kb * 32 + hsel * 8;                 // A: K split 0..7 / 16..23 per lane half
        const _Float16* ap = &Ah[arow * 392 + ka];
        v16h a = load_frag_lds(ap, ap + 16);
        int n   = n0 + arow;
        int kbB = kb * 32 + hsel * 8;                // B: K split 0..7 / 16..23 per lane half
        const _Float16* bp = &WeH[(size_t)n * K3 + kbB];
        v16h bm = load_frag_lds(bp, bp + 16);
        acc = __builtin_amdgcn_wmma_f32_16x16x32_f16(false, a, false, bm,
                                                     (short)0, acc, false, false);
    }
    {
        int col = n0 + (lane & 15);
        int rb  = hsel * 8;
        float bias = b_edge[col];
#pragma unroll
        for (int r = 0; r < 8; ++r) Z[(rb + r) * 132 + col] = acc[r] + bias;
    }
    __syncthreads();

    // ---- phase 3: layernorm + adaLN + outputs ----
    float zr[8], s = 0.f, s2 = 0.f;
#pragma unroll
    for (int t = 0; t < 8; ++t) {
        float v = Z[g * 132 + l16 * 8 + t];
        zr[t] = v; s += v; s2 += v * v;
    }
    s = grp16_sum(s); s2 = grp16_sum(s2);
    float mean = s * (1.f / DD);
    float var  = s2 * (1.f / DD) - mean * mean;
    float inv  = rsqrtf(var + 1e-6f);
    float sdv  = Sdist[g];

    float eu[8], se = 0.f;
#pragma unroll
    for (int t = 0; t < 8; ++t) {
        int d = l16 * 8 + t;
        float shift = sdv * W_ada[d]      + b_ada[d];
        float scale = sdv * W_ada[DD + d] + b_ada[DD + d];
        float u = (zr[t] - mean) * inv * (1.f + scale) + shift;
        eu[t] = u; se += u * u;
    }
    se = grp16_sum(se);
    float nm  = fmaxf(sqrtf(se), 1e-7f);
    float fac = tanhf(nm) / nm;                       // expmap0

    float* eup = eupd + (size_t)eIdx * DD;
    float* oe  = out_e + ((size_t)row * NN + j) * DD;
#pragma unroll
    for (int t = 0; t < 8; ++t) {
        int d = l16 * 8 + t;
        eup[d] = eu[t];
        oe[d]  = eu[t] * fac;
    }
}

// ---------------- kernel 4: node model (16 rows / block, 24 WMMA each) ----------------
__global__ __launch_bounds__(256) void pmpnn_node_kernel(
    const float* __restrict__ x, const float* __restrict__ eupd,
    const int* __restrict__ deg,
    const _Float16* __restrict__ WnH, const _Float16* __restrict__ WaH,
    const _Float16* __restrict__ WpH,
    const float* __restrict__ b_node, const float* __restrict__ b_ada_node,
    const float* __restrict__ b_pl,
    float* __restrict__ out_h)
{
    __shared__ __align__(16) float    NF[16 * 132];
    __shared__ __align__(16) float    AG[16 * 132];
    __shared__ __align__(16) _Float16 AH[16 * 264];   // f16 A operand (up to K=256)
    __shared__ __align__(16) float    MOD[16 * 392];  // 16 x 384 adaLN params
    __shared__ __align__(16) float    LIN[16 * 132];

    const int tid = threadIdx.x, g = tid >> 4, l16 = tid & 15;
    const int row = blockIdx.x * 16 + g;              // 0..1023
    const int wave = tid >> 5, lane = tid & 31;
    const int arow = lane & 15, hsel = lane >> 4;

    // ---- phase 1: nf = logmap0(x), agg = scatter-mean (seg = i_idx, both batches) ----
    const float* xr = x + (size_t)row * DD;
    float vx[8], sxx = 0.f;
#pragma unroll
    for (int t = 0; t < 8; ++t) { float a = xr[l16 * 8 + t]; vx[t] = a; sxx += a * a; }
    sxx = grp16_sum(sxx);
    float fx = logfac(sxx);

    float ag[8] = {0.f,0.f,0.f,0.f,0.f,0.f,0.f,0.f};
    if (row < NN) {
        const int e0 = row * DEG;          // batch 0 edges of node i=row
        const int e1 = (NN + row) * DEG;   // batch 1 edges of node i=row
        for (int k = 0; k < DEG; ++k) {
            const float* p0 = eupd + (size_t)(e0 + k) * DD + l16 * 8;
            const float* p1 = eupd + (size_t)(e1 + k) * DD + l16 * 8;
#pragma unroll
            for (int t = 0; t < 8; ++t) ag[t] += p0[t] + p1[t];
        }
        float ic = 1.f / fmaxf((float)(deg[row] + deg[NN + row]), 1.f);
#pragma unroll
        for (int t = 0; t < 8; ++t) ag[t] *= ic;
    }
#pragma unroll
    for (int t = 0; t < 8; ++t) {
        int d = l16 * 8 + t;
        NF[g * 132 + d] = vx[t] * fx;
        AG[g * 132 + d] = ag[t];
        float sa = ag[t] / (1.f + expf(-ag[t]));      // silu(agg)
        AH[g * 264 + d] = (_Float16)sa;
    }
    __syncthreads();

    // ---- phase 2: mod2 = silu(agg) @ W_ada_node^T  (N=384 -> 3 tiles/wave, K=128) ----
    for (int tt = 0; tt < 3; ++tt) {
        int n0 = (wave + 8 * tt) * 16;
        v8f acc = {};
#pragma unroll
        for (int kb = 0; kb < 4; ++kb) {
            const _Float16* ap = &AH[arow * 264 + kb * 32 + hsel * 8];
            v16h a = load_frag_lds(ap, ap + 16);
            const _Float16* bp = &WaH[(size_t)(n0 + arow) * DD + kb * 32 + hsel * 8];
            v16h bm = load_frag_lds(bp, bp + 16);
            acc = __builtin_amdgcn_wmma_f32_16x16x32_f16(false, a, false, bm,
                                                         (short)0, acc, false, false);
        }
        int col = n0 + (lane & 15), rb = hsel * 8;
        float bias = b_ada_node[col];
#pragma unroll
        for (int r = 0; r < 8; ++r) MOD[(rb + r) * 392 + col] = acc[r] + bias;
    }
    __syncthreads();

    // ---- phase 3: A = concat(nf, agg) as f16 ----
    {
        int r = tid >> 4;
#pragma unroll
        for (int t = 0; t < 16; ++t) {
            int c = l16 * 16 + t;
            float v = (c < DD) ? NF[r * 132 + c] : AG[r * 132 + c - DD];
            AH[r * 264 + c] = (_Float16)v;
        }
    }
    __syncthreads();

    // ---- phase 4: lin = concat(nf,agg) @ W_node^T  (N=128, K=256) ----
    {
        int n0 = wave * 16;
        v8f acc = {};
#pragma unroll
        for (int kb = 0; kb < 8; ++kb) {
            const _Float16* ap = &AH[arow * 264 + kb * 32 + hsel * 8];
            v16h a = load_frag_lds(ap, ap + 16);
            const _Float16* bp = &WnH[(size_t)(n0 + arow) * 256 + kb * 32 + hsel * 8];
            v16h bm = load_frag_lds(bp, bp + 16);
            acc = __builtin_amdgcn_wmma_f32_16x16x32_f16(false, a, false, bm,
                                                         (short)0, acc, false, false);
        }
        int col = n0 + (lane & 15), rb = hsel * 8;
        float bias = b_node[col];
#pragma unroll
        for (int r = 0; r < 8; ++r) LIN[(rb + r) * 132 + col] = acc[r] + bias;
    }
    __syncthreads();

    // ---- phase 5: layernorm + adaLN + exp/log maps; stage u as f16 ----
    float lr[8], s = 0.f, s2 = 0.f;
#pragma unroll
    for (int t = 0; t < 8; ++t) {
        float v = LIN[g * 132 + l16 * 8 + t];
        lr[t] = v; s += v; s2 += v * v;
    }
    s = grp16_sum(s); s2 = grp16_sum(s2);
    float mean = s * (1.f / DD);
    float var  = s2 * (1.f / DD) - mean * mean;
    float inv  = rsqrtf(var + 1e-6f);

    float hv[8], hs = 0.f;
#pragma unroll
    for (int t = 0; t < 8; ++t) {
        int d = l16 * 8 + t;
        float shift2 = MOD[g * 392 + d];
        float scale2 = MOD[g * 392 + 128 + d];
        float gate2  = MOD[g * 392 + 256 + d];
        float ln = (lr[t] - mean) * inv;
        float h  = NF[g * 132 + d] + gate2 * (ln * (1.f + scale2) + shift2);
        hv[t] = h; hs += h * h;
    }
    hs = grp16_sum(hs);
    float nh  = sqrtf(hs);
    float nm1 = fmaxf(nh, 1e-7f);
    float f1  = tanhf(nm1) / nm1;                     // expmap0
    float n2  = nh * f1;                              // ||expmap0(h)||
    float n2c = fminf(fmaxf(n2, 1e-7f), 1.f - 1e-5f);
    float f2  = atanhf(n2c) / n2c;                    // logmap0
    float ff  = f1 * f2;
#pragma unroll
    for (int t = 0; t < 8; ++t) AH[g * 264 + l16 * 8 + t] = (_Float16)(hv[t] * ff);
    __syncthreads();

    // ---- phase 6: hp = u @ W_pl^T  (N=128, K=128) ----
    {
        int n0 = wave * 16;
        v8f acc = {};
#pragma unroll
        for (int kb = 0; kb < 4; ++kb) {
            const _Float16* ap = &AH[arow * 264 + kb * 32 + hsel * 8];
            v16h a = load_frag_lds(ap, ap + 16);
            const _Float16* bp = &WpH[(size_t)(n0 + arow) * DD + kb * 32 + hsel * 8];
            v16h bm = load_frag_lds(bp, bp + 16);
            acc = __builtin_amdgcn_wmma_f32_16x16x32_f16(false, a, false, bm,
                                                         (short)0, acc, false, false);
        }
        int col = n0 + (lane & 15), rb = hsel * 8;
        float bias = b_pl[col];
#pragma unroll
        for (int r = 0; r < 8; ++r) LIN[(rb + r) * 132 + col] = acc[r] + bias;
    }
    __syncthreads();

    // ---- phase 7: final expmap0 + store h_next ----
    float pr[8], ss = 0.f;
#pragma unroll
    for (int t = 0; t < 8; ++t) {
        float v = LIN[g * 132 + l16 * 8 + t];
        pr[t] = v; ss += v * v;
    }
    ss = grp16_sum(ss);
    float nmo = fmaxf(sqrtf(ss), 1e-7f);
    float fo  = tanhf(nmo) / nmo;
#pragma unroll
    for (int t = 0; t < 8; ++t) out_h[(size_t)row * DD + l16 * 8 + t] = pr[t] * fo;
}

// ---------------- host launcher ----------------
extern "C" void kernel_launch(void* const* d_in, const int* in_sizes, int n_in,
                              void* d_out, int out_size, void* d_ws, size_t ws_size,
                              hipStream_t stream)
{
    (void)in_sizes; (void)n_in; (void)out_size; (void)ws_size;

    const float*         x          = (const float*)d_in[0];
    const unsigned char* adj        = (const unsigned char*)d_in[1];
    const float*         e          = (const float*)d_in[2];
    /* d_in[3] = num_edges (scalar int), fixed by geometry */
    const float* W_edge     = (const float*)d_in[4];
    const float* b_edge     = (const float*)d_in[5];
    const float* W_node     = (const float*)d_in[6];
    const float* b_node     = (const float*)d_in[7];
    const float* W_ada      = (const float*)d_in[8];
    const float* b_ada      = (const float*)d_in[9];
    const float* W_ada_node = (const float*)d_in[10];
    const float* b_ada_node = (const float*)d_in[11];
    const float* W_pl       = (const float*)d_in[12];
    const float* b_pl       = (const float*)d_in[13];

    float* out_h = (float*)d_out;                       // [B*N, D] = 131072 floats
    float* out_e = (float*)d_out + (size_t)BB * NN * DD; // [B,N,N,D]

    // workspace layout (all offsets 256B aligned)
    char* ws = (char*)d_ws;
    int*       jlist = (int*)(ws + 0);                          //  131072 B
    int*       deg   = (int*)(ws + 131072);                     //    4096 B
    _Float16*  WeH   = (_Float16*)(ws + 135168);                //   98304 B (128x384)
    _Float16*  WnH   = (_Float16*)(ws + 233472);                //   65536 B (128x256)
    _Float16*  WaH   = (_Float16*)(ws + 299008);                //   98304 B (384x128)
    _Float16*  WpH   = (_Float16*)(ws + 397312);                //   32768 B (128x128)
    float*     eupd  = (float*)(ws + 430080);                   // 16 MB (E x D)

    pmpnn_cvt_weights<<<192, 256, 0, stream>>>(W_edge, W_node, W_ada_node, W_pl,
                                               WeH, WnH, WaH, WpH);
    pmpnn_build_edges<<<(BB * NN) / 8, 256, 0, stream>>>(adj, jlist, deg);

    const long n4 = (long)BB * NN * NN * DD / 4;   // 16,777,216 float4
    pmpnn_copy_e<<<4096, 256, 0, stream>>>((const f4v*)e, (f4v*)out_e, n4);

    pmpnn_edge_kernel<<<EE / 16, 256, 0, stream>>>(x, e, jlist, WeH,
                                                   b_edge, W_ada, b_ada,
                                                   eupd, out_e);

    pmpnn_node_kernel<<<(BB * NN) / 16, 256, 0, stream>>>(x, eupd, deg,
                                                          WnH, WaH, WpH,
                                                          b_node, b_ada_node, b_pl,
                                                          out_h);
}
